// DiffMoeMLP_47562467836362
// MI455X (gfx1250) — matrix operand order repaired and checked by reading, not compiled, not run on hardware
//
#include <hip/hip_runtime.h>

#define BS_TOK 16384
#define D_DIM  1024
#define DD_DIM 4096
#define NEXP   8
#define KCAP   2048
#define LNEPS  1e-5f
#define LSTR2  72   // LDS tile row stride in 16-bit elems for K=64 tiles (pad vs banks)

typedef unsigned short u16;
typedef __attribute__((ext_vector_type(16))) __bf16 v16bf;
typedef __attribute__((ext_vector_type(8)))  float  v8f;
typedef __attribute__((ext_vector_type(8)))  u16    ush8;

union FragU { v16bf v; ush8 h[2]; };

__device__ __forceinline__ u16 f2bf(float f) {
  unsigned u = __float_as_uint(f);
  u += 0x7FFFu + ((u >> 16) & 1u);       // round-to-nearest-even
  return (u16)(u >> 16);
}

__device__ __forceinline__ float wred32(float v) {
#pragma unroll
  for (int o = 16; o > 0; o >>= 1) v += __shfl_xor(v, o, 32);
  return v;
}

__device__ __forceinline__ float gelu_tanh(float x) {
  float u = 0.7978845608f * (x + 0.044715f * x * x * x);
  return 0.5f * x * (1.0f + tanhf(u));
}

// Workgroup barrier that drains ONLY DScnt (LDS) before signaling, leaving
// in-flight global loads (LOADcnt) unwaited so they overlap the WMMA phase.
// Safe for the LDS double-buffer protocol: every thread's LDS reads/writes
// complete before it signals; buffer WAR hazards are one full barrier apart.
__device__ __forceinline__ void lds_only_barrier() {
  asm volatile(
      "s_wait_dscnt 0x0\n\t"
      "s_barrier_signal -1\n\t"
      "s_barrier_wait -1" ::: "memory");
}

// ---------------------------------------------------------------------------
// Kernel 1: scores = softmax(x @ gate_w^T)   (16384 tokens x 8 experts)
// ---------------------------------------------------------------------------
__global__ __launch_bounds__(256) void moe_gate_kernel(
    const float* __restrict__ x, const float* __restrict__ gate_w,
    float* __restrict__ scores) {
  __shared__ float sg[NEXP * D_DIM];                  // 32 KB
  for (int i = threadIdx.x; i < NEXP * D_DIM; i += 256) sg[i] = gate_w[i];
  __syncthreads();

  const int wave = threadIdx.x >> 5, lane = threadIdx.x & 31;
  const int t = blockIdx.x * 8 + wave;
  const float* xr = x + (size_t)t * D_DIM;

  float acc[NEXP] = {};
  for (int d = lane; d < D_DIM; d += 32) {
    float xv = xr[d];
#pragma unroll
    for (int e = 0; e < NEXP; ++e) acc[e] = fmaf(xv, sg[e * D_DIM + d], acc[e]);
  }
#pragma unroll
  for (int e = 0; e < NEXP; ++e) acc[e] = wred32(acc[e]);

  if (lane == 0) {
    float m = acc[0];
#pragma unroll
    for (int e = 1; e < NEXP; ++e) m = fmaxf(m, acc[e]);
    float ex[NEXP], s = 0.f;
#pragma unroll
    for (int e = 0; e < NEXP; ++e) { ex[e] = __expf(acc[e] - m); s += ex[e]; }
    float inv = 1.f / s;
#pragma unroll
    for (int e = 0; e < NEXP; ++e) scores[(size_t)t * NEXP + e] = ex[e] * inv;
  }
}

// ---------------------------------------------------------------------------
// Kernel 2: exact per-expert top-KCAP selection (set semantics; order free).
// ---------------------------------------------------------------------------
__global__ __launch_bounds__(256) void moe_topk_kernel(
    const float* __restrict__ scores, int* __restrict__ idx) {
  const int e = blockIdx.x;
  __shared__ int s_cnt;
  __shared__ unsigned s_lo, s_hi;
  __shared__ int s_pos;

  if (threadIdx.x == 0) { s_lo = 0u; s_hi = 0x3F800001u; }  // scores in [0,1]
  __syncthreads();

  for (int it = 0; it < 32; ++it) {
    unsigned lo = s_lo, hi = s_hi;
    unsigned mid = lo + ((hi - lo) >> 1);
    if (mid == lo) break;                    // uniform across block
    if (threadIdx.x == 0) s_cnt = 0;
    __syncthreads();
    int c = 0;
    for (int t = threadIdx.x; t < BS_TOK; t += 256)
      if (__float_as_uint(scores[(size_t)t * NEXP + e]) >= mid) ++c;
    atomicAdd(&s_cnt, c);
    __syncthreads();
    if (threadIdx.x == 0) { if (s_cnt >= KCAP) s_lo = mid; else s_hi = mid; }
    __syncthreads();
  }
  __syncthreads();
  const unsigned tau = s_lo;   // count(>=tau) >= KCAP, count(>tau) < KCAP

  if (threadIdx.x == 0) s_pos = 0;
  __syncthreads();
  for (int t = threadIdx.x; t < BS_TOK; t += 256) {
    if (__float_as_uint(scores[(size_t)t * NEXP + e]) > tau) {
      int p = atomicAdd(&s_pos, 1);
      idx[e * KCAP + p] = t;
    }
  }
  __syncthreads();
  if (threadIdx.x == 0) {                    // deterministic tie fill
    int p = s_pos;
    for (int t = 0; t < BS_TOK && p < KCAP; ++t)
      if (__float_as_uint(scores[(size_t)t * NEXP + e]) == tau)
        idx[e * KCAP + p++] = t;
  }
}

// ---------------------------------------------------------------------------
// Kernel 3: gather selected tokens, LayerNorm, convert to bf16 -> Y.
// ---------------------------------------------------------------------------
__global__ __launch_bounds__(256) void moe_gather_ln_kernel(
    const float* __restrict__ x, const int* __restrict__ idx,
    const float* __restrict__ scores, const float* __restrict__ ln_w,
    const float* __restrict__ ln_b, u16* __restrict__ Y,
    float* __restrict__ gw) {
  const int wave = threadIdx.x >> 5, lane = threadIdx.x & 31;
  const int rid = blockIdx.x * 8 + wave;        // 0 .. 16383
  const int e = rid >> 11;                      // rid / KCAP
  const int t = idx[rid];
  const float* xr = x + (size_t)t * D_DIM;

  float v[32], sum = 0.f, sq = 0.f;
#pragma unroll
  for (int j = 0; j < 32; ++j) {
    float f = xr[lane + 32 * j];
    v[j] = f; sum += f; sq = fmaf(f, f, sq);
  }
  sum = wred32(sum); sq = wred32(sq);
  const float mean = sum * (1.f / D_DIM);
  const float var  = sq * (1.f / D_DIM) - mean * mean;
  const float rstd = rsqrtf(var + LNEPS);

  u16* yr = Y + (size_t)rid * D_DIM;
#pragma unroll
  for (int j = 0; j < 32; ++j) {
    int d = lane + 32 * j;
    float yv = (v[j] - mean) * rstd * ln_w[d] + ln_b[d];
    yr[d] = f2bf(yv);
  }
  if (lane == 0) gw[rid] = scores[(size_t)t * NEXP + e];
}

// ---------------------------------------------------------------------------
// GEMM helpers: block tile 128x128, K-step 64, double-buffered LDS,
// wave grid 2x4, each wave owns 64x32 as 4x2 WMMA accumulators.
// ---------------------------------------------------------------------------
__device__ __forceinline__ void gloadA(const u16* __restrict__ g, int ldg,
                                       int k0, ush8 r[4]) {
  const int row = threadIdx.x >> 1;
  const int cb = (threadIdx.x & 1) * 32;
#pragma unroll
  for (int j = 0; j < 4; ++j)
    r[j] = *(const ush8*)(g + (size_t)row * ldg + k0 + cb + j * 8);
}

__device__ __forceinline__ void gloadB(const float* __restrict__ g, int ldg,
                                       int k0, float4 r[8]) {
  const int row = threadIdx.x >> 1;
  const int cb = (threadIdx.x & 1) * 32;
#pragma unroll
  for (int j = 0; j < 8; ++j)
    r[j] = *(const float4*)(g + (size_t)row * ldg + k0 + cb + j * 4);
}

__device__ __forceinline__ void storeA(u16* sA, const ush8 r[4]) {
  const int row = threadIdx.x >> 1;
  const int cb = (threadIdx.x & 1) * 32;
#pragma unroll
  for (int j = 0; j < 4; ++j)
    *(ush8*)(sA + row * LSTR2 + cb + j * 8) = r[j];
}

__device__ __forceinline__ void storeB(u16* sB, const float4 r[8]) {
  const int row = threadIdx.x >> 1;
  const int cb = (threadIdx.x & 1) * 32;
#pragma unroll
  for (int j = 0; j < 8; j += 2) {
    ush8 v;
    v[0] = f2bf(r[j].x);     v[1] = f2bf(r[j].y);
    v[2] = f2bf(r[j].z);     v[3] = f2bf(r[j].w);
    v[4] = f2bf(r[j + 1].x); v[5] = f2bf(r[j + 1].y);
    v[6] = f2bf(r[j + 1].z); v[7] = f2bf(r[j + 1].w);
    *(ush8*)(sB + row * LSTR2 + cb + j * 4) = v;
  }
}

// A fragment (ISA 7.12.2, 16-bit A 16x32): lane<16 row M=lane, K=kb..kb+7 and
// kb+16..kb+23 with kb=0; lanes 16..31 kb=8.
__device__ __forceinline__ v16bf frag_A(const u16* sA, int row0, int kc) {
  const int lane = threadIdx.x & 31;
  const u16* p = sA + (row0 + (lane & 15)) * LSTR2 + kc + ((lane >> 4) * 8);
  FragU u;
  u.h[0] = *(const ush8*)p;
  u.h[1] = *(const ush8*)(p + 16);
  return u.v;
}

// B fragment: lane column N = lane&15, K = kb..kb+7 and kb+16..kb+23, kb=(lane>>4)*8.
__device__ __forceinline__ v16bf frag_B(const u16* sB, int row0, int kc) {
  const int lane = threadIdx.x & 31;
  const u16* p = sB + (row0 + (lane & 15)) * LSTR2 + kc + ((lane >> 4) * 8);
  FragU u;
  u.h[0] = *(const ush8*)p;
  u.h[1] = *(const ush8*)(p + 16);
  return u.v;
}

// ---------------------------------------------------------------------------
// Kernel 4: H[e] = gelu(Y[e] @ fc1[e]^T + b1[e])   M=2048 N=4096 K=1024
// ---------------------------------------------------------------------------
__global__ __launch_bounds__(256, 2) void moe_gemm1_kernel(
    const u16* __restrict__ Y, const float* __restrict__ fc1s,
    const float* __restrict__ b1s, u16* __restrict__ H) {
  __shared__ __attribute__((aligned(16))) u16 sA[2][128 * LSTR2];
  __shared__ __attribute__((aligned(16))) u16 sB[2][128 * LSTR2];

  const int e = blockIdx.z;
  const int blockM = blockIdx.y * 128;
  const int blockN = blockIdx.x * 128;
  const u16*   A = Y    + (size_t)e * KCAP * D_DIM + (size_t)blockM * D_DIM;
  const float* W = fc1s + (size_t)e * DD_DIM * D_DIM + (size_t)blockN * D_DIM;

  const int wave = threadIdx.x >> 5, lane = threadIdx.x & 31;
  const int wm = wave >> 2, wn = wave & 3;

  v8f acc[4][2];
#pragma unroll
  for (int mi = 0; mi < 4; ++mi)
#pragma unroll
    for (int ni = 0; ni < 2; ++ni)
      acc[mi][ni] = (v8f){0.f, 0.f, 0.f, 0.f, 0.f, 0.f, 0.f, 0.f};

  ush8 ra[4];
  float4 rb[8];
  gloadA(A, D_DIM, 0, ra);
  gloadB(W, D_DIM, 0, rb);

  const int nk = D_DIM / 64;
  for (int kt = 0; kt < nk; ++kt) {
    const int buf = kt & 1;
    storeA(sA[buf], ra);
    storeB(sB[buf], rb);
    if (kt + 1 < nk) {
      gloadA(A, D_DIM, (kt + 1) * 64, ra);
      gloadB(W, D_DIM, (kt + 1) * 64, rb);
      if (kt + 2 < nk)   // next-next weight tile, WGP scope prefetch
        __builtin_prefetch(
            (const void*)(W + (size_t)(threadIdx.x >> 1) * D_DIM + (kt + 2) * 64), 0, 3);
    }
    lds_only_barrier();   // drain DScnt only; global loads stay in flight
#pragma unroll
    for (int c = 0; c < 2; ++c) {
      v16bf af[4], bfr[2];
#pragma unroll
      for (int mi = 0; mi < 4; ++mi) af[mi] = frag_A(sA[buf], wm * 64 + mi * 16, c * 32);
#pragma unroll
      for (int ni = 0; ni < 2; ++ni) bfr[ni] = frag_B(sB[buf], wn * 32 + ni * 16, c * 32);
#pragma unroll
      for (int mi = 0; mi < 4; ++mi)
#pragma unroll
        for (int ni = 0; ni < 2; ++ni)
          acc[mi][ni] = __builtin_amdgcn_wmma_f32_16x16x32_bf16(
              false, af[mi], false, bfr[ni], (short)0, acc[mi][ni], false, false);
    }
  }

  const float* b1 = b1s + (size_t)e * DD_DIM;
#pragma unroll
  for (int mi = 0; mi < 4; ++mi)
#pragma unroll
    for (int ni = 0; ni < 2; ++ni) {
      int n = blockN + wn * 32 + ni * 16 + (lane & 15);
      float bias = b1[n];
      int mbase = blockM + wm * 64 + mi * 16 + ((lane >> 4) * 8);
#pragma unroll
      for (int r = 0; r < 8; ++r) {
        int m = mbase + r;
        float vv = gelu_tanh(acc[mi][ni][r] + bias);
        H[((size_t)e * KCAP + m) * DD_DIM + n] = f2bf(vv);
      }
    }
}

// ---------------------------------------------------------------------------
// Kernel 5: out[t] += gate * (H[e] @ fc2[e]^T + b2[e])  M=2048 N=1024 K=4096
// ---------------------------------------------------------------------------
__global__ __launch_bounds__(256, 2) void moe_gemm2_kernel(
    const u16* __restrict__ H, const float* __restrict__ fc2s,
    const float* __restrict__ b2s, const int* __restrict__ idx,
    const float* __restrict__ gw, float* __restrict__ out) {
  __shared__ __attribute__((aligned(16))) u16 sA[2][128 * LSTR2];
  __shared__ __attribute__((aligned(16))) u16 sB[2][128 * LSTR2];

  const int e = blockIdx.z;
  const int blockM = blockIdx.y * 128;
  const int blockN = blockIdx.x * 128;
  const u16*   A = H    + (size_t)e * KCAP * DD_DIM + (size_t)blockM * DD_DIM;
  const float* W = fc2s + (size_t)e * D_DIM * DD_DIM + (size_t)blockN * DD_DIM;

  const int wave = threadIdx.x >> 5, lane = threadIdx.x & 31;
  const int wm = wave >> 2, wn = wave & 3;

  v8f acc[4][2];
#pragma unroll
  for (int mi = 0; mi < 4; ++mi)
#pragma unroll
    for (int ni = 0; ni < 2; ++ni)
      acc[mi][ni] = (v8f){0.f, 0.f, 0.f, 0.f, 0.f, 0.f, 0.f, 0.f};

  ush8 ra[4];
  float4 rb[8];
  gloadA(A, DD_DIM, 0, ra);
  gloadB(W, DD_DIM, 0, rb);

  const int nk = DD_DIM / 64;
  for (int kt = 0; kt < nk; ++kt) {
    const int buf = kt & 1;
    storeA(sA[buf], ra);
    storeB(sB[buf], rb);
    if (kt + 1 < nk) {
      gloadA(A, DD_DIM, (kt + 1) * 64, ra);
      gloadB(W, DD_DIM, (kt + 1) * 64, rb);
      if (kt + 2 < nk)
        __builtin_prefetch(
            (const void*)(W + (size_t)(threadIdx.x >> 1) * DD_DIM + (kt + 2) * 64), 0, 3);
    }
    lds_only_barrier();
#pragma unroll
    for (int c = 0; c < 2; ++c) {
      v16bf af[4], bfr[2];
#pragma unroll
      for (int mi = 0; mi < 4; ++mi) af[mi] = frag_A(sA[buf], wm * 64 + mi * 16, c * 32);
#pragma unroll
      for (int ni = 0; ni < 2; ++ni) bfr[ni] = frag_B(sB[buf], wn * 32 + ni * 16, c * 32);
#pragma unroll
      for (int mi = 0; mi < 4; ++mi)
#pragma unroll
        for (int ni = 0; ni < 2; ++ni)
          acc[mi][ni] = __builtin_amdgcn_wmma_f32_16x16x32_bf16(
              false, af[mi], false, bfr[ni], (short)0, acc[mi][ni], false, false);
    }
  }

  const float* b2 = b2s + (size_t)e * D_DIM;
#pragma unroll
  for (int mi = 0; mi < 4; ++mi)
#pragma unroll
    for (int ni = 0; ni < 2; ++ni) {
      int n = blockN + wn * 32 + ni * 16 + (lane & 15);
      float bias = b2[n];
      int mbase = blockM + wm * 64 + mi * 16 + ((lane >> 4) * 8);
#pragma unroll
      for (int r = 0; r < 8; ++r) {
        int m = mbase + r;
        int row = e * KCAP + m;
        int t = idx[row];
        float g = gw[row];
        float vv = (acc[mi][ni][r] + bias) * g;
        atomicAdd(out + (size_t)t * D_DIM + n, vv);   // global_atomic_add_f32
      }
    }
}

// ---------------------------------------------------------------------------
// Host launcher
// ---------------------------------------------------------------------------
extern "C" void kernel_launch(void* const* d_in, const int* in_sizes, int n_in,
                              void* d_out, int out_size, void* d_ws, size_t ws_size,
                              hipStream_t stream) {
  const float* x      = (const float*)d_in[0];
  const float* gate_w = (const float*)d_in[1];
  const float* ln_w   = (const float*)d_in[2];
  const float* ln_b   = (const float*)d_in[3];
  const float* fc1s   = (const float*)d_in[4];
  const float* b1s    = (const float*)d_in[5];
  const float* fc2s   = (const float*)d_in[6];
  const float* b2s    = (const float*)d_in[7];
  float* out = (float*)d_out;

  char* ws = (char*)d_ws;
  float* scores = (float*)ws;                                   // 512 KB
  int*   idx    = (int*)(ws + (size_t)512 * 1024);              //  64 KB
  float* gw     = (float*)(ws + (size_t)576 * 1024);            //  64 KB
  u16*   Y      = (u16*)(ws + (size_t)640 * 1024);              //  32 MB
  u16*   Hbuf   = (u16*)(ws + (size_t)640 * 1024 +
                         (size_t)NEXP * KCAP * D_DIM * 2);      // 128 MB

  // out starts as a copy of x; gemm2 scatter-adds expert outputs into it.
  hipMemcpyAsync(out, x, (size_t)BS_TOK * D_DIM * sizeof(float),
                 hipMemcpyDeviceToDevice, stream);

  moe_gate_kernel<<<BS_TOK / 8, 256, 0, stream>>>(x, gate_w, scores);
  moe_topk_kernel<<<NEXP, 256, 0, stream>>>(scores, idx);
  moe_gather_ln_kernel<<<(NEXP * KCAP) / 8, 256, 0, stream>>>(
      x, idx, scores, ln_w, ln_b, Y, gw);
  moe_gemm1_kernel<<<dim3(DD_DIM / 128, KCAP / 128, NEXP), 256, 0, stream>>>(
      Y, fc1s, b1s, Hbuf);
  moe_gemm2_kernel<<<dim3(D_DIM / 128, KCAP / 128, NEXP), 256, 0, stream>>>(
      Hbuf, fc2s, b2s, idx, gw, out);
}
